// input_masked_conv2d_44392781971789
// MI455X (gfx1250) — matrix operand order repaired and checked by reading, not compiled, not run on hardware
//
#include <hip/hip_runtime.h>

// ---------------- problem constants ----------------
#define CIN   128
#define COUT  128
#define HH    32
#define WW    32
#define NTAPS 9
#define KTOT  (CIN * NTAPS)     // 1152
#define CCH   32                // channels per chunk
#define NCC   (CIN / CCH)       // 4
#define NTILE 128               // pixels per WG tile (4 rows of 32)

// LDS geometry
#define XROWS    6              // 4 rows + halo
#define XRSTRIDE 40             // floats per x row; data at [4..35], halo cols [3],[36]
#define WTSTRIDE 296            // ushorts per weight row (288 + 8 pad, 16B aligned)
#define BSTRIDE  40             // ushorts per B row (32 + 8 pad, 16B aligned)

// repacked weight in workspace: [cc][m][kk*32+cl] bf16, row = 288 bf16 = 144 uints
#define WP_UINTS_PER_ROW 144
#define WP_TOTAL_UINTS   (NCC * COUT * WP_UINTS_PER_ROW)   // 73728 uints = 294912 B

typedef __attribute__((ext_vector_type(16))) __bf16          v16bf;
typedef __attribute__((ext_vector_type(8)))  float           v8f;
typedef __attribute__((ext_vector_type(8)))  unsigned short  v8u;
typedef __attribute__((ext_vector_type(4)))  float           v4f;
typedef __attribute__((ext_vector_type(4)))  unsigned int    v4ui;

union FragU {
    v8u   h[2];
    v16bf bf;
};

// Pack two fp32 -> bf16x2 (low = a, high = b), RNE.
#if __has_builtin(__builtin_amdgcn_cvt_pk_bf16_f32)
typedef __attribute__((ext_vector_type(2))) __bf16 v2bf;
static __device__ __forceinline__ unsigned int pk_bf16(float a, float b) {
    v2bf r = __builtin_amdgcn_cvt_pk_bf16_f32(a, b);
    return __builtin_bit_cast(unsigned int, r);
}
#else
static __device__ __forceinline__ unsigned int pk_bf16(float a, float b) {
    unsigned int ua = __builtin_bit_cast(unsigned int, a);
    unsigned int ub = __builtin_bit_cast(unsigned int, b);
    ua += 0x7fffu + ((ua >> 16) & 1u);
    ub += 0x7fffu + ((ub >> 16) & 1u);
    return (ua >> 16) | (ub & 0xffff0000u);
}
#endif

// ---- CDNA5 async global->LDS copy (ASYNCcnt), per ISA 15.18.3 ----
static __device__ __forceinline__ unsigned lds_off_u32(const void* p) {
    // generic shared pointer: low 32 bits = LDS byte address
    return (unsigned)(unsigned long long)p;
}
static __device__ __forceinline__ void async_g2l_b128(unsigned lds, const void* g) {
    unsigned long long ga = (unsigned long long)g;
    asm volatile("global_load_async_to_lds_b128 %0, %1, off"
                 :: "v"(lds), "v"(ga) : "memory");
}
static __device__ __forceinline__ void wait_async0() {
    asm volatile("s_wait_asynccnt 0x0" ::: "memory");
}

// ================= pre-pass: repack weights fp32 -> bf16 tap-major =================
__global__ __launch_bounds__(256)
void repack_weight_kernel(const float* __restrict__ weight, unsigned int* __restrict__ wp) {
    const int d = blockIdx.x * 256 + threadIdx.x;      // uint index, 0..73727
    const int rm  = d % WP_UINTS_PER_ROW;              // kk*16 + clp
    const int row = d / WP_UINTS_PER_ROW;              // cc*128 + m
    const int clp = rm & 15;
    const int kk  = rm >> 4;
    const int m   = row & 127;
    const int cc  = row >> 7;
    const int c   = cc * CCH + clp * 2;
    const float a = weight[m * KTOT + c * NTAPS + kk];
    const float b = weight[m * KTOT + (c + 1) * NTAPS + kk];
    wp[d] = pk_bf16(a, b);
}

// ================= main kernel =================
template <bool USE_WS>
__global__ __launch_bounds__(256)
void masked_conv_wmma_kernel(const float* __restrict__ x,
                             const float* __restrict__ maskg,
                             const float* __restrict__ weight,
                             const float* __restrict__ bias,
                             const unsigned int* __restrict__ wp,
                             float* __restrict__ out)
{
    __shared__ alignas(16) float          XLds[CCH * XROWS * XRSTRIDE];
    __shared__ alignas(16) unsigned short WtLds[COUT * WTSTRIDE];
    __shared__ alignas(16) unsigned short BLds[2][NTILE * BSTRIDE];    // double buffered
    __shared__ float maskLds[NTAPS * NTILE];
    __shared__ float biasLds[COUT];

    const int t    = threadIdx.x;
    const int lane = t & 31;
    const int wave = t >> 5;
    const int wm   = wave & 3;
    const int wn   = wave >> 2;
    const int kh   = lane >> 4;
    const int l15  = lane & 15;

    const int blk  = blockIdx.x;
    const int bimg = blk >> 3;
    const int h0   = (blk & 7) * 4;
    const int xbase = bimg * CIN * HH * WW;

    for (int idx = t; idx < NTAPS * NTILE; idx += 256) {
        const int kk = idx >> 7;
        const int n  = idx & 127;
        maskLds[idx] = maskg[kk * (HH * WW) + h0 * WW + n];
    }
    if (t < COUT) biasLds[t] = bias[t];

    v8f acc[2][4];
#pragma unroll
    for (int i = 0; i < 2; ++i)
#pragma unroll
        for (int j = 0; j < 4; ++j)
            acc[i][j] = (v8f)(0.0f);

    for (int cc = 0; cc < NCC; ++cc) {
        const int c0 = cc * CCH;

        // ---------- stage X halo tile (async global->LDS), zero padding ----------
        if (t < CCH * XROWS) {
            const int cl = t / XROWS;
            const int r  = t - cl * XROWS;
            const int h  = h0 - 1 + r;
            float* dst = &XLds[(cl * XROWS + r) * XRSTRIDE];
            if ((unsigned)h < (unsigned)HH) {
                const float* src = x + xbase + (c0 + cl) * (HH * WW) + h * WW;
                const unsigned ldst = lds_off_u32(dst + 4);
#pragma unroll
                for (int q = 0; q < 8; ++q)
                    async_g2l_b128(ldst + q * 16, src + q * 4);
            } else {
#pragma unroll
                for (int q = 0; q < 8; ++q)
                    *reinterpret_cast<v4f*>(dst + 4 + q * 4) = (v4f)(0.0f);
            }
            dst[3]  = 0.0f;
            dst[36] = 0.0f;
        }

        // ---------- stage Wt ----------
        if constexpr (USE_WS) {
            // dense copy of pre-repacked bf16 (async global->LDS)
            const int m    = t >> 1;
            const int half = t & 1;
            const unsigned int* src = wp + (cc * COUT + m) * WP_UINTS_PER_ROW + half * 72;
            const unsigned ldst = lds_off_u32(&WtLds[m * WTSTRIDE + half * 144]);
#pragma unroll
            for (int i = 0; i < 18; ++i)
                async_g2l_b128(ldst + i * 16, src + i * 4);
        } else {
            // in-kernel conversion fallback
            const int m  = t >> 1;
            const int jb = (t & 1) * 16;
            const float* wsrc = weight + m * KTOT + c0 * NTAPS + jb * NTAPS;
            unsigned short* wdst = &WtLds[m * WTSTRIDE + jb];
#pragma unroll
            for (int clp = 0; clp < 8; ++clp) {
#pragma unroll
                for (int kk = 0; kk < NTAPS; ++kk) {
                    const float a = wsrc[clp * 18 + kk];
                    const float b = wsrc[clp * 18 + 9 + kk];
                    *reinterpret_cast<unsigned int*>(&wdst[kk * CCH + clp * 2]) = pk_bf16(a, b);
                }
            }
        }

        wait_async0();       // drain ASYNCcnt before making LDS visible
        __syncthreads();     // X + Wt ready

#pragma unroll
        for (int kk = 0; kk < NTAPS; ++kk) {
            const int di  = kk / 3 - 1;
            const int dj  = kk % 3 - 1;
            unsigned short* Bbuf = BLds[kk & 1];

            // ---------- build B tile (branch-free LDS->LDS) ----------
            {
                const int n   = t & 127;
                const int ks  = (t >> 7) * 16;
                const int r   = 1 + (n >> 5) + di;
                const int col = 4 + (n & 31) + dj;
                const float mk = maskLds[kk * NTILE + n];
                const float* xsrc = &XLds[(ks * XROWS + r) * XRSTRIDE + col];
                v4ui q0, q1;
#pragma unroll
                for (int e2 = 0; e2 < 4; ++e2) {
                    const float a = xsrc[(2 * e2)     * (XROWS * XRSTRIDE)] * mk;
                    const float b = xsrc[(2 * e2 + 1) * (XROWS * XRSTRIDE)] * mk;
                    q0[e2] = pk_bf16(a, b);
                }
#pragma unroll
                for (int e2 = 0; e2 < 4; ++e2) {
                    const float a = xsrc[(8 + 2 * e2) * (XROWS * XRSTRIDE)] * mk;
                    const float b = xsrc[(9 + 2 * e2) * (XROWS * XRSTRIDE)] * mk;
                    q1[e2] = pk_bf16(a, b);
                }
                *reinterpret_cast<v4ui*>(&Bbuf[n * BSTRIDE + ks])     = q0;
                *reinterpret_cast<v4ui*>(&Bbuf[n * BSTRIDE + ks + 8]) = q1;
            }

            __syncthreads();   // single barrier per tap (double-buffered B)

            FragU af[2], bfr[4];
            const int kb = kk * CCH;
#pragma unroll
            for (int ms = 0; ms < 2; ++ms) {
                const int m = wm * 32 + ms * 16 + l15;
                const unsigned short* wr = &WtLds[m * WTSTRIDE + kb];
                af[ms].h[0] = *reinterpret_cast<const v8u*>(wr + kh * 8);
                af[ms].h[1] = *reinterpret_cast<const v8u*>(wr + 16 + kh * 8);
            }
#pragma unroll
            for (int ns = 0; ns < 4; ++ns) {
                const int n = wn * 64 + ns * 16 + l15;
                const unsigned short* br = &Bbuf[n * BSTRIDE + kh * 8];
                bfr[ns].h[0] = *reinterpret_cast<const v8u*>(br);
                bfr[ns].h[1] = *reinterpret_cast<const v8u*>(br + 16);
            }

#pragma unroll
            for (int ms = 0; ms < 2; ++ms)
#pragma unroll
                for (int ns = 0; ns < 4; ++ns)
                    acc[ms][ns] = __builtin_amdgcn_wmma_f32_16x16x32_bf16(
                        false, af[ms].bf, false, bfr[ns].bf,
                        (short)0, acc[ms][ns], false, false);
        }

        __syncthreads();       // protect XLds/WtLds/BLds before next chunk's staging
    }

    // ---------- epilogue ----------
    const int obase = bimg * COUT * HH * WW;
#pragma unroll
    for (int ms = 0; ms < 2; ++ms) {
#pragma unroll
        for (int ns = 0; ns < 4; ++ns) {
            const int n = wn * 64 + ns * 16 + l15;
            const int h = h0 + (n >> 5);
            const int w = n & 31;
#pragma unroll
            for (int r = 0; r < 8; ++r) {
                const int o = wm * 32 + ms * 16 + r + 8 * kh;
                out[obase + (o * HH + h) * WW + w] = acc[ms][ns][r] + biasLds[o];
            }
        }
    }
}

extern "C" void kernel_launch(void* const* d_in, const int* in_sizes, int n_in,
                              void* d_out, int out_size, void* d_ws, size_t ws_size,
                              hipStream_t stream) {
    const float* x      = (const float*)d_in[0];
    const float* maskg  = (const float*)d_in[1];
    const float* weight = (const float*)d_in[2];
    const float* bias   = (const float*)d_in[3];
    float* out = (float*)d_out;

    dim3 grid(32 * 8);
    dim3 block(256);

    if (ws_size >= (size_t)WP_TOTAL_UINTS * 4) {
        unsigned int* wp = (unsigned int*)d_ws;
        repack_weight_kernel<<<dim3(WP_TOTAL_UINTS / 256), block, 0, stream>>>(weight, wp);
        masked_conv_wmma_kernel<true><<<grid, block, 0, stream>>>(x, maskg, weight, bias, wp, out);
    } else {
        masked_conv_wmma_kernel<false><<<grid, block, 0, stream>>>(x, maskg, weight, bias, nullptr, out);
    }
}
